// SelfAttention_73186242724010
// MI455X (gfx1250) — compile-verified, matrix operand order checked by reading, not run on hardware
//
#include <hip/hip_runtime.h>
#include <math.h>

// ---------------------------------------------------------------------------
// Problem constants (from reference)
// ---------------------------------------------------------------------------
#define BB   4
#define SS   2048
#define DD   1024
#define HH   16
#define DH   64
#define ND3  (3 * DD)          // 3072
#define MROWS (BB * SS)        // 8192
#define ATT_SCALE 0.125f       // 1/sqrt(64)

typedef __attribute__((ext_vector_type(16))) __bf16 v16bf;
typedef __attribute__((ext_vector_type(8)))  __bf16 v8bf;
typedef __attribute__((ext_vector_type(8)))  float  v8f;

union V16U { v16bf v; v8bf h[2]; };

// Load a 16-element bf16 A/B operand fragment from two 16-byte chunks.
// ISA 16-bit operand layout: lane holds row (lane&15); kbase = 8*(lane>>4);
// elements 0..7 = K kbase..kbase+7, elements 8..15 = K kbase+16..kbase+23.
static __device__ inline v16bf ld16(const __bf16* p0, const __bf16* p1) {
    V16U u;
    u.h[0] = *(const v8bf*)p0;
    u.h[1] = *(const v8bf*)p1;
    return u.v;
}

static __device__ inline v8f wmma_bf16(v16bf a, v16bf b, v8f c) {
    return __builtin_amdgcn_wmma_f32_16x16x32_bf16(
        /*neg_a=*/false, a, /*neg_b=*/false, b,
        /*c_mod=*/(short)0, c, /*reuse_a=*/false, /*reuse_b=*/false);
}

// ---------------------------------------------------------------------------
// Prep kernels: fp32 -> bf16 (plain and transposed)
// ---------------------------------------------------------------------------
__global__ void f2bf_kernel(const float* __restrict__ src,
                            __bf16* __restrict__ dst, int n) {
    int i = blockIdx.x * blockDim.x + threadIdx.x;
    if (i < n) dst[i] = (__bf16)src[i];
}

// src is [K][N] row-major; dst is [N][K] row-major (i.e. transposed)
__global__ void f2bf_t_kernel(const float* __restrict__ src,
                              __bf16* __restrict__ dst, int K, int N) {
    int i = blockIdx.x * blockDim.x + threadIdx.x;
    if (i < K * N) {
        int k = i / N;
        int n = i - k * N;
        dst[(size_t)n * K + k] = (__bf16)src[i];
    }
}

// ---------------------------------------------------------------------------
// QKV GEMM: [8192 x 1024] @ [1024 x 3072] + bias, scatter into Q/K (B,H,S,DH)
// and transposed Vt (B,H,DH,S).  One wave computes a 32x64 output tile:
// per K=32 step -> 2 A loads + 4 B loads feed 8 WMMAs.
// ---------------------------------------------------------------------------
__global__ __launch_bounds__(128)
void qkv_gemm_kernel(const __bf16* __restrict__ xb,
                     const __bf16* __restrict__ wt,    // [3072][1024] = W_in^T
                     const float*  __restrict__ bias,  // [3072]
                     __bf16* __restrict__ qb,
                     __bf16* __restrict__ kb,
                     __bf16* __restrict__ vtb) {
    int wave  = (blockIdx.x * blockDim.x + threadIdx.x) >> 5;
    int lane  = threadIdx.x & 31;
    int half  = lane >> 4;
    int l15   = lane & 15;
    int kbase = half * 8;

    const int tilesM = MROWS / 32;       // 256
    int tm = wave % tilesM;
    int tn = wave / tilesM;              // 0..47
    int mbase = tm * 32;
    int nbase = tn * 64;

    const __bf16* arow0 = xb + (size_t)(mbase + l15) * DD;
    const __bf16* arow1 = xb + (size_t)(mbase + 16 + l15) * DD;

    v8f acc[2][4] = {};
#pragma unroll 2
    for (int kk = 0; kk < DD; kk += 32) {
        v16bf a0 = ld16(arow0 + kk + kbase, arow0 + kk + kbase + 16);
        v16bf a1 = ld16(arow1 + kk + kbase, arow1 + kk + kbase + 16);
        __builtin_prefetch(arow0 + kk + 256, 0, 1);
        __builtin_prefetch(arow1 + kk + 256, 0, 1);
#pragma unroll
        for (int j = 0; j < 4; ++j) {
            const __bf16* brow = wt + (size_t)(nbase + j * 16 + l15) * DD;
            v16bf b = ld16(brow + kk + kbase, brow + kk + kbase + 16);
            acc[0][j] = wmma_bf16(a0, b, acc[0][j]);
            acc[1][j] = wmma_bf16(a1, b, acc[1][j]);
        }
    }

#pragma unroll
    for (int j = 0; j < 4; ++j) {
        int n     = nbase + j * 16 + l15;
        int which = n >> 10;             // 0=q 1=k 2=v
        int d     = n & (DD - 1);
        int h     = d >> 6;
        int dh    = d & (DH - 1);
        float bv  = bias[n];
#pragma unroll
        for (int i = 0; i < 2; ++i) {
#pragma unroll
            for (int r = 0; r < 8; ++r) {
                int m = mbase + i * 16 + r + half * 8;
                int b = m >> 11;         // /2048
                int s = m & (SS - 1);
                float val = acc[i][j][r] + bv;
                int bh = b * HH + h;
                if (which == 0) {
                    qb[((size_t)bh * SS + s) * DH + dh] = (__bf16)(val * ATT_SCALE);
                } else if (which == 1) {
                    kb[((size_t)bh * SS + s) * DH + dh] = (__bf16)val;
                } else {
                    vtb[((size_t)bh * DH + dh) * SS + s] = (__bf16)val;
                }
            }
        }
    }
}

// ---------------------------------------------------------------------------
// Flash attention: one wave handles 32 query rows of one (b,h).
// Online softmax over 64 key-chunks of 32; P tiles staged through LDS to
// convert C-layout -> A-layout for the PV matmul.
// Per chunk: 8 score WMMAs + 8 PV WMMAs against 8 global tile loads.
// ---------------------------------------------------------------------------
__global__ __launch_bounds__(128)
void attn_kernel(const __bf16* __restrict__ qg,
                 const __bf16* __restrict__ kg,
                 const __bf16* __restrict__ vtg,
                 __bf16* __restrict__ og) {      // [B*S][D]
    __shared__ __bf16 plds[4][32 * 32];          // 2 KB per wave

    int wave  = (blockIdx.x * blockDim.x + threadIdx.x) >> 5;
    int wslot = (threadIdx.x >> 5);
    int lane  = threadIdx.x & 31;
    int half  = lane >> 4;
    int l15   = lane & 15;
    int kbase = half * 8;

    int qt = wave & 63;                  // query tile 0..63 (32 rows each)
    int bh = wave >> 6;                  // 0..63
    int qbase = qt * 32;

    const __bf16* Q  = qg  + (size_t)bh * SS * DH;
    const __bf16* K  = kg  + (size_t)bh * SS * DH;
    const __bf16* Vt = vtg + (size_t)bh * DH * SS;

    // Q A-fragments: [row tile i][K=32 step over DH] (Q already scaled)
    v16bf qa[2][2];
#pragma unroll
    for (int i = 0; i < 2; ++i) {
        const __bf16* qrow = Q + (size_t)(qbase + i * 16 + l15) * DH;
        qa[i][0] = ld16(qrow + kbase,      qrow + kbase + 16);
        qa[i][1] = ld16(qrow + 32 + kbase, qrow + 32 + kbase + 16);
    }

    float mrow[2][8], lrow[2][8];
#pragma unroll
    for (int i = 0; i < 2; ++i)
#pragma unroll
        for (int r = 0; r < 8; ++r) { mrow[i][r] = -3.0e38f; lrow[i][r] = 0.0f; }
    v8f acc[2][4] = {};

    __bf16* pl = &plds[wslot][0];

    for (int c = 0; c < SS; c += 32) {
        // ---- scores: 2 row tiles x 2 key tiles, contraction over DH=64 ----
        v8f s[2][2] = {};
        const __bf16* kr0 = K + (size_t)(c + l15) * DH;
        const __bf16* kr1 = K + (size_t)(c + 16 + l15) * DH;
        v16bf b00 = ld16(kr0 + kbase,      kr0 + kbase + 16);
        v16bf b01 = ld16(kr0 + 32 + kbase, kr0 + 32 + kbase + 16);
        v16bf b10 = ld16(kr1 + kbase,      kr1 + kbase + 16);
        v16bf b11 = ld16(kr1 + 32 + kbase, kr1 + 32 + kbase + 16);
#pragma unroll
        for (int i = 0; i < 2; ++i) {
            s[i][0] = wmma_bf16(qa[i][0], b00, s[i][0]);
            s[i][0] = wmma_bf16(qa[i][1], b01, s[i][0]);
            s[i][1] = wmma_bf16(qa[i][0], b10, s[i][1]);
            s[i][1] = wmma_bf16(qa[i][1], b11, s[i][1]);
        }

        // ---- online softmax update (row lives across the 16 lanes of a half)
#pragma unroll
        for (int i = 0; i < 2; ++i) {
#pragma unroll
            for (int r = 0; r < 8; ++r) {
                float mx = fmaxf(s[i][0][r], s[i][1][r]);
                mx = fmaxf(mx, __shfl_xor(mx, 1, 32));
                mx = fmaxf(mx, __shfl_xor(mx, 2, 32));
                mx = fmaxf(mx, __shfl_xor(mx, 4, 32));
                mx = fmaxf(mx, __shfl_xor(mx, 8, 32));
                float mn = fmaxf(mrow[i][r], mx);
                float sc = __expf(mrow[i][r] - mn);
                mrow[i][r] = mn;
                float p0 = __expf(s[i][0][r] - mn);
                float p1 = __expf(s[i][1][r] - mn);
                float ps = p0 + p1;
                ps += __shfl_xor(ps, 1, 32);
                ps += __shfl_xor(ps, 2, 32);
                ps += __shfl_xor(ps, 4, 32);
                ps += __shfl_xor(ps, 8, 32);
                lrow[i][r] = lrow[i][r] * sc + ps;
                acc[i][0][r] *= sc; acc[i][1][r] *= sc;
                acc[i][2][r] *= sc; acc[i][3][r] *= sc;
                int m = i * 16 + r + half * 8;
                pl[m * 32 + l15]      = (__bf16)p0;   // keys c..c+15
                pl[m * 32 + 16 + l15] = (__bf16)p1;   // keys c+16..c+31
            }
        }

        // LDS store->load within wave; DS ops are in-order, force the wait.
        asm volatile("s_wait_dscnt 0" ::: "memory");

        // ---- reload P as A operands (two 16x32 tiles) ----
        v16bf pa[2];
#pragma unroll
        for (int i = 0; i < 2; ++i) {
            const __bf16* prow = pl + (i * 16 + l15) * 32;
            pa[i] = ld16(prow + kbase, prow + kbase + 16);
        }

        // ---- O += P @ V  (B operand rows contiguous thanks to Vt layout) ----
#pragma unroll
        for (int j = 0; j < 4; ++j) {
            const __bf16* vrow = Vt + (size_t)(j * 16 + l15) * SS;
            v16bf vb = ld16(vrow + c + kbase, vrow + c + kbase + 16);
            acc[0][j] = wmma_bf16(pa[0], vb, acc[0][j]);
            acc[1][j] = wmma_bf16(pa[1], vb, acc[1][j]);
        }
    }

    // ---- normalize and store to [B*S][D] bf16 ----
    int b = bh >> 4;
    int h = bh & (HH - 1);
#pragma unroll
    for (int i = 0; i < 2; ++i) {
#pragma unroll
        for (int r = 0; r < 8; ++r) {
            float inv = 1.0f / lrow[i][r];
            int s = qbase + i * 16 + r + half * 8;
            size_t row = (size_t)(b * SS + s) * DD + h * DH;
#pragma unroll
            for (int j = 0; j < 4; ++j) {
                og[row + j * 16 + l15] = (__bf16)(acc[i][j][r] * inv);
            }
        }
    }
}

// ---------------------------------------------------------------------------
// Output projection: [8192 x 1024] @ [1024 x 1024] + b_out -> fp32 out
// One wave computes a 32x64 tile.
// ---------------------------------------------------------------------------
__global__ __launch_bounds__(128)
void out_gemm_kernel(const __bf16* __restrict__ ab,   // [8192][1024]
                     const __bf16* __restrict__ wt,   // [1024][1024] = W_out^T
                     const float*  __restrict__ bias, // [1024]
                     float* __restrict__ out) {       // [8192][1024]
    int wave  = (blockIdx.x * blockDim.x + threadIdx.x) >> 5;
    int lane  = threadIdx.x & 31;
    int half  = lane >> 4;
    int l15   = lane & 15;
    int kbase = half * 8;

    const int tilesM = MROWS / 32;       // 256
    int tm = wave % tilesM;
    int tn = wave / tilesM;              // 0..15
    int mbase = tm * 32;
    int nbase = tn * 64;

    const __bf16* arow0 = ab + (size_t)(mbase + l15) * DD;
    const __bf16* arow1 = ab + (size_t)(mbase + 16 + l15) * DD;

    v8f acc[2][4] = {};
#pragma unroll 2
    for (int kk = 0; kk < DD; kk += 32) {
        v16bf a0 = ld16(arow0 + kk + kbase, arow0 + kk + kbase + 16);
        v16bf a1 = ld16(arow1 + kk + kbase, arow1 + kk + kbase + 16);
        __builtin_prefetch(arow0 + kk + 256, 0, 1);
        __builtin_prefetch(arow1 + kk + 256, 0, 1);
#pragma unroll
        for (int j = 0; j < 4; ++j) {
            const __bf16* brow = wt + (size_t)(nbase + j * 16 + l15) * DD;
            v16bf b = ld16(brow + kk + kbase, brow + kk + kbase + 16);
            acc[0][j] = wmma_bf16(a0, b, acc[0][j]);
            acc[1][j] = wmma_bf16(a1, b, acc[1][j]);
        }
    }

#pragma unroll
    for (int j = 0; j < 4; ++j) {
        int n = nbase + j * 16 + l15;
        float bv = bias[n];
#pragma unroll
        for (int i = 0; i < 2; ++i) {
#pragma unroll
            for (int r = 0; r < 8; ++r) {
                int m = mbase + i * 16 + r + half * 8;
                out[(size_t)m * DD + n] = acc[i][j][r] + bv;
            }
        }
    }
}

// ---------------------------------------------------------------------------
// Launch
// ---------------------------------------------------------------------------
extern "C" void kernel_launch(void* const* d_in, const int* in_sizes, int n_in,
                              void* d_out, int out_size, void* d_ws, size_t ws_size,
                              hipStream_t stream) {
    const float* x     = (const float*)d_in[0];   // [B,S,D]
    const float* W_in  = (const float*)d_in[1];   // [D,3D]
    const float* b_in  = (const float*)d_in[2];   // [3D]
    const float* W_out = (const float*)d_in[3];   // [D,D]
    const float* b_out = (const float*)d_in[4];   // [D]
    float* out = (float*)d_out;

    char* ws = (char*)d_ws;
    size_t off = 0;
    auto carve = [&](size_t bytes) {
        void* p = ws + off;
        off += (bytes + 255) & ~(size_t)255;
        return p;
    };
    __bf16* xb    = (__bf16*)carve((size_t)MROWS * DD * 2);   // 16 MB
    __bf16* wqkvt = (__bf16*)carve((size_t)ND3 * DD * 2);     //  6 MB
    __bf16* wot   = (__bf16*)carve((size_t)DD * DD * 2);      //  2 MB
    __bf16* qb    = (__bf16*)carve((size_t)MROWS * DD * 2);   // 16 MB
    __bf16* kb    = (__bf16*)carve((size_t)MROWS * DD * 2);   // 16 MB
    __bf16* vtb   = (__bf16*)carve((size_t)MROWS * DD * 2);   // 16 MB
    __bf16* attn  = (__bf16*)carve((size_t)MROWS * DD * 2);   // 16 MB

    int nx = MROWS * DD;                 // 8388608
    f2bf_kernel<<<(nx + 255) / 256, 256, 0, stream>>>(x, xb, nx);
    int nwi = DD * ND3;                  // 3145728
    f2bf_t_kernel<<<(nwi + 255) / 256, 256, 0, stream>>>(W_in, wqkvt, DD, ND3);
    int nwo = DD * DD;                   // 1048576
    f2bf_t_kernel<<<(nwo + 255) / 256, 256, 0, stream>>>(W_out, wot, DD, DD);

    // QKV: 256 x 48 tiles of 32x64, 4 waves/block
    qkv_gemm_kernel<<<(256 * 48) / 4, 128, 0, stream>>>(xb, wqkvt, b_in, qb, kb, vtb);

    // Attention: 64 (b,h) x 64 query tiles (32 rows each) = 4096 waves
    attn_kernel<<<4096 / 4, 128, 0, stream>>>(qb, kb, vtb, attn);

    // Out projection: 256 x 16 tiles of 32x64
    out_gemm_kernel<<<(256 * 16) / 4, 128, 0, stream>>>(attn, wot, b_out, out);
}
